// TransformerDecoder_75050258530497
// MI455X (gfx1250) — compile-verified, matrix operand order checked by reading, not run on hardware
//
#include <hip/hip_runtime.h>
#include <hip/hip_bf16.h>

#define LAYERS 6
#define DMODEL 1024
#define FDIM   4096
#define NHEAD  16
#define BATCH  2
#define SEQ    2048
#define HDIM   64
#define TOK    (BATCH * SEQ)   /* 4096 token rows */
#define EPS    1e-5f

typedef __attribute__((ext_vector_type(16))) __bf16 v16bf;
typedef __attribute__((ext_vector_type(8)))  __bf16 v8bf;
typedef __attribute__((ext_vector_type(8)))  float  v8f;

static __device__ __forceinline__ v16bf join16(v8bf lo, v8bf hi) {
  v16bf r;
#pragma unroll
  for (int j = 0; j < 8; ++j) { r[j] = lo[j]; r[8 + j] = hi[j]; }
  return r;
}

// CDNA5 async copy: global memory -> LDS, 16 bytes per lane, no VGPR data.
// Generic LDS pointer's low 32 bits are the LDS byte address (ISA 10.2),
// so we can pass it straight to the instruction.
static __device__ __forceinline__ void async_cp16(const __bf16* g, __bf16* l) {
  unsigned int  lo = (unsigned int)(unsigned long long)(uintptr_t)l;
  unsigned long long ga = (unsigned long long)(uintptr_t)g;
  asm volatile("global_load_async_to_lds_b128 %0, %1, off"
               :: "v"(lo), "v"(ga)
               : "memory");
}
static __device__ __forceinline__ void async_wait0() {
  asm volatile("s_wait_asynccnt 0x0" ::: "memory");
}

// ---------------------------------------------------------------- converts
__global__ __launch_bounds__(256)
void k_cvt_bf16(const float* __restrict__ in, __bf16* __restrict__ out, int n) {
  int i = blockIdx.x * 256 + threadIdx.x;
  if (i < n) out[i] = (__bf16)in[i];
}

// in: K x N (row major fp32)  ->  out: N x K (row major bf16)
__global__ __launch_bounds__(256)
void k_cvt_tr_bf16(const float* __restrict__ in, __bf16* __restrict__ out,
                   int K, int N) {
  int i = blockIdx.x * 256 + threadIdx.x;
  if (i < K * N) {
    int k = i / N, n = i - k * N;
    out[(size_t)n * K + k] = (__bf16)in[i];
  }
}

// ---------------------------------------------------------------- GEMM
// C[M,N] = A[M,K] (bf16) * Bt[N,K]^T (bf16) + bias[N]  (+ optional ReLU)
// 128x128 block tile, 8 waves (4 in M x 2 in N), wave tile 32x64 (2x4 WMMA).
// Double-buffered LDS tiles filled with global_load_async_to_lds_b128.
__global__ __launch_bounds__(256)
void k_gemm_bf16(const __bf16* __restrict__ A, const __bf16* __restrict__ Bt,
                 const float* __restrict__ bias, float* __restrict__ C,
                 int M, int Nd, int K, int relu) {
  __shared__ __bf16 As[2][128][40];   // 40*2B = 80B row stride (16B aligned)
  __shared__ __bf16 Bs[2][128][40];

  const int tid  = threadIdx.x;
  const int w    = tid >> 5;
  const int lane = tid & 31;
  const int ln   = lane & 15;
  const int hi   = lane >> 4;
  const int kb   = hi * 8;          // K base within fragment per ISA layout
  const int wm   = w & 3;           // wave row 0..3  (32 rows each)
  const int wn   = w >> 2;          // wave col 0..1  (64 cols each)
  const int mb   = blockIdx.y * 128;
  const int nb   = blockIdx.x * 128;

  const int lrow = tid >> 1;        // 0..127
  const int lk   = (tid & 1) * 16;  // 0 or 16

  v8f acc[2][4];
#pragma unroll
  for (int i = 0; i < 2; ++i)
#pragma unroll
    for (int j = 0; j < 4; ++j) acc[i][j] = (v8f)0.f;

  const __bf16* aptr = A  + (size_t)(mb + lrow) * K + lk;
  const __bf16* bptr = Bt + (size_t)(nb + lrow) * K + lk;

  // stage tile for k0 into buffer buf (4 async b128 copies per thread)
  auto stage = [&](int buf, int k0) {
    async_cp16(aptr + k0,     &As[buf][lrow][lk]);
    async_cp16(aptr + k0 + 8, &As[buf][lrow][lk + 8]);
    async_cp16(bptr + k0,     &Bs[buf][lrow][lk]);
    async_cp16(bptr + k0 + 8, &Bs[buf][lrow][lk + 8]);
  };

  stage(0, 0);
  async_wait0();
  __syncthreads();

  int cur = 0;
  for (int k0 = 0; k0 < K; k0 += 32) {
    if (k0 + 32 < K) stage(cur ^ 1, k0 + 32);   // prefetch next tile

    v16bf a[2], b[4];
#pragma unroll
    for (int i = 0; i < 2; ++i) {
      int m = wm * 32 + i * 16 + ln;
      a[i] = join16(*(const v8bf*)&As[cur][m][kb],
                    *(const v8bf*)&As[cur][m][16 + kb]);
    }
#pragma unroll
    for (int j = 0; j < 4; ++j) {
      int n = wn * 64 + j * 16 + ln;
      b[j] = join16(*(const v8bf*)&Bs[cur][n][kb],
                    *(const v8bf*)&Bs[cur][n][16 + kb]);
    }
#pragma unroll
    for (int i = 0; i < 2; ++i)
#pragma unroll
      for (int j = 0; j < 4; ++j)
        acc[i][j] = __builtin_amdgcn_wmma_f32_16x16x32_bf16(
            false, a[i], false, b[j], (short)0, acc[i][j], false, false);

    async_wait0();     // my prefetch landed
    __syncthreads();   // everyone's prefetch landed / reads done
    cur ^= 1;
  }

#pragma unroll
  for (int i = 0; i < 2; ++i) {
#pragma unroll
    for (int j = 0; j < 4; ++j) {
      int col = nb + wn * 64 + j * 16 + ln;
      float bv = bias ? bias[col] : 0.f;
#pragma unroll
      for (int r = 0; r < 8; ++r) {
        int row = mb + wm * 32 + i * 16 + hi * 8 + r;   // C/D layout: m=r+8*hi
        float v = acc[i][j][r] + bv;
        if (relu) v = v > 0.f ? v : 0.f;
        C[(size_t)row * Nd + col] = v;
      }
    }
  }
}

// ---------------------------------------------------------------- attention
// one block per (b, h, 64-query tile); flash-style online softmax, causal
__global__ __launch_bounds__(256)
void k_attention(const float* __restrict__ qkv, float* __restrict__ y) {
  __shared__ __bf16 Qs[64][72];
  __shared__ __bf16 Ks[64][72];
  __shared__ __bf16 VsT[64][72];   // [head-dim][key]
  __shared__ __bf16 Ps[64][72];
  __shared__ float  Stile[64][66];
  __shared__ float  row_max[64], row_sum[64], row_scale[64];

  const int tid  = threadIdx.x;
  const int w    = tid >> 5;
  const int lane = tid & 31;
  const int ln   = lane & 15;
  const int hi   = lane >> 4;
  const int kb   = hi * 8;
  const int qt   = blockIdx.x;
  const int h    = blockIdx.y;
  const int b    = blockIdx.z;
  const int q0   = qt * 64;
  const size_t tokbase = (size_t)b * SEQ;

  v8f acc[2];
  acc[0] = (v8f)0.f;
  acc[1] = (v8f)0.f;

  { // load + scale Q tile (1/sqrt(64) = 0.125)
    int r  = tid >> 2;
    int d0 = (tid & 3) * 16;
    const float* src = qkv + (tokbase + q0 + r) * (3 * DMODEL) + h * HDIM + d0;
#pragma unroll
    for (int j = 0; j < 16; ++j) Qs[r][d0 + j] = (__bf16)(src[j] * 0.125f);
  }
  if (tid < 64) { row_max[tid] = -1e30f; row_sum[tid] = 0.f; }
  __syncthreads();

  for (int kt = 0; kt <= qt; ++kt) {
    const int k0 = kt * 64;
    { // load K row-major, V transposed
      int r  = tid >> 2;
      int d0 = (tid & 3) * 16;
      const float* ksrc =
          qkv + (tokbase + k0 + r) * (3 * DMODEL) + DMODEL + h * HDIM + d0;
      const float* vsrc = ksrc + DMODEL;
#pragma unroll
      for (int j = 0; j < 16; ++j) {
        Ks[r][d0 + j]  = (__bf16)ksrc[j];
        VsT[d0 + j][r] = (__bf16)vsrc[j];
      }
    }
    __syncthreads();

    // S = Q K^T : 16 16x16 tiles, 2 per wave, K=64 as two x32 WMMAs
#pragma unroll
    for (int t = 0; t < 2; ++t) {
      int idx = w + t * 8;
      int ti = idx >> 2, tj = idx & 3;
      v8f s = (v8f)0.f;
#pragma unroll
      for (int ks = 0; ks < 2; ++ks) {
        int ko = ks * 32;
        int m = ti * 16 + ln;
        int n = tj * 16 + ln;
        v16bf a  = join16(*(const v8bf*)&Qs[m][ko + kb],
                          *(const v8bf*)&Qs[m][ko + 16 + kb]);
        v16bf bb = join16(*(const v8bf*)&Ks[n][ko + kb],
                          *(const v8bf*)&Ks[n][ko + 16 + kb]);
        s = __builtin_amdgcn_wmma_f32_16x16x32_bf16(false, a, false, bb,
                                                    (short)0, s, false, false);
      }
#pragma unroll
      for (int r = 0; r < 8; ++r)
        Stile[ti * 16 + hi * 8 + r][tj * 16 + ln] = s[r];
    }
    __syncthreads();

    // online softmax, one thread per query row; causal mask j <= q-k0
    if (tid < 64) {
      const int r = tid;
      const int jlim = q0 + r - k0;
      const int jend = jlim < 63 ? jlim : 63;
      float mprev = row_max[r];
      float mnew  = mprev;
      for (int j = 0; j <= jend; ++j) {
        float sv = Stile[r][j];
        mnew = sv > mnew ? sv : mnew;
      }
      float scl = __expf(mprev - mnew);
      float sum = 0.f;
      for (int j = 0; j < 64; ++j) {
        float p = (j <= jend) ? __expf(Stile[r][j] - mnew) : 0.f;
        Ps[r][j] = (__bf16)p;
        sum += p;
      }
      row_sum[r]   = row_sum[r] * scl + sum;
      row_max[r]   = mnew;
      row_scale[r] = scl;
    }
    __syncthreads();

    // O = O*scale + P V
#pragma unroll
    for (int t = 0; t < 2; ++t) {
      int idx = w + t * 8;
      int ti = idx >> 2, tj = idx & 3;
#pragma unroll
      for (int r = 0; r < 8; ++r) acc[t][r] *= row_scale[ti * 16 + hi * 8 + r];
#pragma unroll
      for (int ks = 0; ks < 2; ++ks) {
        int ko = ks * 32;
        int m = ti * 16 + ln;
        int n = tj * 16 + ln;
        v16bf a  = join16(*(const v8bf*)&Ps[m][ko + kb],
                          *(const v8bf*)&Ps[m][ko + 16 + kb]);
        v16bf bb = join16(*(const v8bf*)&VsT[n][ko + kb],
                          *(const v8bf*)&VsT[n][ko + 16 + kb]);
        acc[t] = __builtin_amdgcn_wmma_f32_16x16x32_bf16(
            false, a, false, bb, (short)0, acc[t], false, false);
      }
    }
    __syncthreads();
  }

  // finalize: divide by row sum, scatter to y[b][n][h*S+s]
#pragma unroll
  for (int t = 0; t < 2; ++t) {
    int idx = w + t * 8;
    int ti = idx >> 2, tj = idx & 3;
#pragma unroll
    for (int r = 0; r < 8; ++r) {
      int m = ti * 16 + hi * 8 + r;
      float ov = acc[t][r] / row_sum[m];
      y[(tokbase + q0 + m) * DMODEL + h * HDIM + tj * 16 + ln] = ov;
    }
  }
}

// ---------------------------------------------------------------- residual+LN
__global__ __launch_bounds__(256)
void k_resid_ln(const float* __restrict__ x, const float* __restrict__ hbuf,
                const float* __restrict__ g, const float* __restrict__ be,
                float* __restrict__ out) {
  __shared__ float red[256];
  const int tid = threadIdx.x;
  const size_t base = (size_t)blockIdx.x * DMODEL;
  float v[4];
#pragma unroll
  for (int i = 0; i < 4; ++i) {
    int c = tid + i * 256;
    v[i] = x[base + c] + hbuf[base + c];
  }
  red[tid] = v[0] + v[1] + v[2] + v[3];
  __syncthreads();
  for (int st = 128; st > 0; st >>= 1) {
    if (tid < st) red[tid] += red[tid + st];
    __syncthreads();
  }
  float mu = red[0] * (1.0f / DMODEL);
  __syncthreads();
  float vs = 0.f;
#pragma unroll
  for (int i = 0; i < 4; ++i) { float d = v[i] - mu; vs += d * d; }
  red[tid] = vs;
  __syncthreads();
  for (int st = 128; st > 0; st >>= 1) {
    if (tid < st) red[tid] += red[tid + st];
    __syncthreads();
  }
  float rstd = rsqrtf(red[0] * (1.0f / DMODEL) + EPS);
#pragma unroll
  for (int i = 0; i < 4; ++i) {
    int c = tid + i * 256;
    out[base + c] = (v[i] - mu) * rstd * g[c] + be[c];
  }
}

// ---------------------------------------------------------------- launcher
extern "C" void kernel_launch(void* const* d_in, const int* in_sizes, int n_in,
                              void* d_out, int out_size, void* d_ws,
                              size_t ws_size, hipStream_t stream) {
  (void)in_sizes; (void)n_in; (void)out_size; (void)ws_size;
  const float* x    = (const float*)d_in[0];
  const float* wqkv = (const float*)d_in[1];
  const float* bqkv = (const float*)d_in[2];
  const float* wo   = (const float*)d_in[3];
  const float* bo   = (const float*)d_in[4];
  const float* w1   = (const float*)d_in[5];
  const float* b1   = (const float*)d_in[6];
  const float* w2   = (const float*)d_in[7];
  const float* b2   = (const float*)d_in[8];
  const float* g1   = (const float*)d_in[9];
  const float* be1  = (const float*)d_in[10];
  const float* g2   = (const float*)d_in[11];
  const float* be2  = (const float*)d_in[12];
  float* out = (float*)d_out;

  char* ws = (char*)d_ws;
  size_t off = 0;
  float*  x_cur = (float*)(ws + off); off += (size_t)TOK * DMODEL * 4;
  float*  qkvb  = (float*)(ws + off); off += (size_t)TOK * 3 * DMODEL * 4;
  float*  yb    = (float*)(ws + off); off += (size_t)TOK * DMODEL * 4;
  float*  hffn  = (float*)(ws + off); off += (size_t)TOK * FDIM * 4;
  float*  tmp   = (float*)(ws + off); off += (size_t)TOK * DMODEL * 4;
  __bf16* actb  = (__bf16*)(ws + off); off += (size_t)TOK * FDIM * 2;
  __bf16* wT    = (__bf16*)(ws + off); off += (size_t)FDIM * DMODEL * 2;

  hipMemcpyAsync(x_cur, x, (size_t)TOK * DMODEL * 4,
                 hipMemcpyDeviceToDevice, stream);

  const dim3 blk(256);
  for (int l = 0; l < LAYERS; ++l) {
    // ---- qkv = x @ wqkv + bqkv
    k_cvt_bf16<<<TOK * DMODEL / 256, blk, 0, stream>>>(x_cur, actb,
                                                       TOK * DMODEL);
    k_cvt_tr_bf16<<<DMODEL * 3 * DMODEL / 256, blk, 0, stream>>>(
        wqkv + (size_t)l * DMODEL * 3 * DMODEL, wT, DMODEL, 3 * DMODEL);
    k_gemm_bf16<<<dim3(3 * DMODEL / 128, TOK / 128), blk, 0, stream>>>(
        actb, wT, bqkv + (size_t)l * 3 * DMODEL, qkvb, TOK, 3 * DMODEL, DMODEL,
        0);

    // ---- fused causal attention -> yb
    k_attention<<<dim3(SEQ / 64, NHEAD, BATCH), blk, 0, stream>>>(qkvb, yb);

    // ---- y @ wo + bo -> tmp ; x = LN(x + tmp)
    k_cvt_bf16<<<TOK * DMODEL / 256, blk, 0, stream>>>(yb, actb, TOK * DMODEL);
    k_cvt_tr_bf16<<<DMODEL * DMODEL / 256, blk, 0, stream>>>(
        wo + (size_t)l * DMODEL * DMODEL, wT, DMODEL, DMODEL);
    k_gemm_bf16<<<dim3(DMODEL / 128, TOK / 128), blk, 0, stream>>>(
        actb, wT, bo + (size_t)l * DMODEL, tmp, TOK, DMODEL, DMODEL, 0);
    k_resid_ln<<<TOK, blk, 0, stream>>>(x_cur, tmp, g1 + (size_t)l * DMODEL,
                                        be1 + (size_t)l * DMODEL, x_cur);

    // ---- h = relu(x @ w1 + b1)
    k_cvt_bf16<<<TOK * DMODEL / 256, blk, 0, stream>>>(x_cur, actb,
                                                       TOK * DMODEL);
    k_cvt_tr_bf16<<<DMODEL * FDIM / 256, blk, 0, stream>>>(
        w1 + (size_t)l * DMODEL * FDIM, wT, DMODEL, FDIM);
    k_gemm_bf16<<<dim3(FDIM / 128, TOK / 128), blk, 0, stream>>>(
        actb, wT, b1 + (size_t)l * FDIM, hffn, TOK, FDIM, DMODEL, 1);

    // ---- h @ w2 + b2 -> tmp ; x = LN(x + tmp)
    k_cvt_bf16<<<TOK * FDIM / 256, blk, 0, stream>>>(hffn, actb, TOK * FDIM);
    k_cvt_tr_bf16<<<FDIM * DMODEL / 256, blk, 0, stream>>>(
        w2 + (size_t)l * FDIM * DMODEL, wT, FDIM, DMODEL);
    k_gemm_bf16<<<dim3(DMODEL / 128, TOK / 128), blk, 0, stream>>>(
        actb, wT, b2 + (size_t)l * DMODEL, tmp, TOK, DMODEL, FDIM, 0);
    float* lnout = (l == LAYERS - 1) ? out : x_cur;
    k_resid_ln<<<TOK, blk, 0, stream>>>(x_cur, tmp, g2 + (size_t)l * DMODEL,
                                        be2 + (size_t)l * DMODEL, lnout);
  }
}
